// _Sam2HieraBlock_22797686407801
// MI455X (gfx1250) — compile-verified, matrix operand order checked by reading, not run on hardware
//
#include <hip/hip_runtime.h>
#include <hip/hip_bf16.h>
#include <math.h>

// ---------------------------------------------------------------------------
// GPT-2 style transformer block for MI455X (gfx1250), bf16 WMMA everywhere.
// B=4, S=2048, D=768, H=12, hd=64, DFF=3072.
// Weights are pre-transposed to bf16 [N][K] so every LDS tile is staged with
// contiguous 128-bit copies -> GLOBAL_LOAD_ASYNC_TO_LDS_B128 (ASYNCcnt).
// ---------------------------------------------------------------------------

typedef __attribute__((ext_vector_type(16))) __bf16 v16bf;
typedef __attribute__((ext_vector_type(8)))  float  v8f;
typedef __attribute__((ext_vector_type(4)))  int    v4i;

#define WMMA_BF16(a, b, c) \
    __builtin_amdgcn_wmma_f32_16x16x32_bf16(false, (a), false, (b), (short)0, (c), false, false)

#if defined(__AMDGCN__) && __has_builtin(__builtin_amdgcn_global_load_async_to_lds_b128)
#define HAS_ASYNC 1
#else
#define HAS_ASYNC 0
#endif

__device__ __forceinline__ __bf16 f2bf(float x) {
    union { float f; unsigned u; } v; v.f = x;
    unsigned r = v.u + 0x7FFFu + ((v.u >> 16) & 1u);   // round-to-nearest-even
    unsigned short h = (unsigned short)(r >> 16);
    return __builtin_bit_cast(__bf16, h);
}

// Stage 16 bytes global -> LDS. Async (ASYNCcnt) when the toolchain exposes
// the gfx1250 builtin, else a plain register round-trip copy.
__device__ __forceinline__ void stage_b128(const __bf16* g, __bf16* l) {
#if HAS_ASYNC
    typedef __attribute__((address_space(1))) v4i gv4i;
    typedef __attribute__((address_space(3))) v4i lv4i;
    __builtin_amdgcn_global_load_async_to_lds_b128(
        (gv4i*)(void*)(const_cast<__bf16*>(g)),
        (lv4i*)(void*)(l), 0, 0);
#else
    *(uint4*)l = *(const uint4*)g;
#endif
}

__device__ __forceinline__ void async_join() {
#if HAS_ASYNC
#if __has_builtin(__builtin_amdgcn_s_wait_asynccnt)
    __builtin_amdgcn_s_wait_asynccnt(0);
#else
    asm volatile("s_wait_asynccnt 0x0" ::: "memory");
#endif
#endif
}

// Load a 16-lane WMMA operand fragment (A-pattern) from LDS.
// lane L: row = L%16 ; elements 0..7 -> K = (L/16)*8 + 0..7,
//         elements 8..15 -> K = 16 + (L/16)*8 + 0..7   (ISA 7.12.2)
// B fragments use the same pattern with row->column, so B tiles are staged
// K-contiguous / N-major (hence the pre-transposed weights).
__device__ __forceinline__ v16bf ld_frag(const __bf16* tile, int stride, int lane) {
    const __bf16* p = tile + (size_t)(lane & 15) * stride + ((lane >> 4) << 3);
    union { v16bf v; uint4 q[2]; } f;
    f.q[0] = *(const uint4*)(p);
    f.q[1] = *(const uint4*)(p + 16);
    return f.v;
}

// ---------------------------------------------------------------------------
// Weight prep: f32 [K][N]  ->  bf16 [N][K]   (32x32 LDS-tiled transpose)
// ---------------------------------------------------------------------------
__global__ __launch_bounds__(256) void transpose_bf16_kernel(const float* __restrict__ in,
                                                             __bf16* __restrict__ out,
                                                             int K, int N) {
    __shared__ float tile[32][33];
    const int tx = threadIdx.x & 31;
    const int ty = threadIdx.x >> 5;          // 8 rows per pass
    const int k0 = blockIdx.y * 32, n0 = blockIdx.x * 32;
#pragma unroll
    for (int r = ty; r < 32; r += 8)
        tile[r][tx] = in[(size_t)(k0 + r) * N + n0 + tx];
    __syncthreads();
#pragma unroll
    for (int r = ty; r < 32; r += 8)
        out[(size_t)(n0 + r) * K + k0 + tx] = f2bf(tile[tx][r]);
}

// ---------------------------------------------------------------------------
// LayerNorm: one 256-thread block per row of D=768, bf16 output.
// ---------------------------------------------------------------------------
__global__ __launch_bounds__(256) void layernorm_kernel(const float* __restrict__ x,
                                                        const float* __restrict__ g,
                                                        const float* __restrict__ b,
                                                        __bf16* __restrict__ out, int D) {
    int row = blockIdx.x;
    const float* xr = x + (size_t)row * D;
    float s = 0.f, ss = 0.f;
    for (int i = threadIdx.x; i < D; i += 256) { float v = xr[i]; s += v; ss += v * v; }
    __shared__ float rs[256], rq[256];
    rs[threadIdx.x] = s; rq[threadIdx.x] = ss;
    __syncthreads();
    for (int off = 128; off > 0; off >>= 1) {
        if (threadIdx.x < off) { rs[threadIdx.x] += rs[threadIdx.x + off];
                                 rq[threadIdx.x] += rq[threadIdx.x + off]; }
        __syncthreads();
    }
    float mu  = rs[0] / (float)D;
    float var = rq[0] / (float)D - mu * mu;
    float inv = rsqrtf(var + 1e-6f);
    for (int i = threadIdx.x; i < D; i += 256)
        out[(size_t)row * D + i] = f2bf((xr[i] - mu) * inv * g[i] + b[i]);
}

// ---------------------------------------------------------------------------
// bf16 WMMA GEMM:  C[M,N] = A[M,K] * Bt[N,K]^T  (+ bias + templated epilogue)
// Block tile 128x128, BK=64, 256 threads = 8 waves; wave tile 64x32 (4x2 frags).
// Both LDS tiles are contiguous 128-bit async copies.
// ---------------------------------------------------------------------------
enum { EP_QKV = 0, EP_ATTN_RES = 1, EP_GELU = 2, EP_OUT_RES = 3 };

template <int MODE>
__global__ __launch_bounds__(256) void gemm_bf16_kernel(
    const __bf16* __restrict__ A, const __bf16* __restrict__ Bt,
    const float*  __restrict__ bias, const float* __restrict__ res,
    void* __restrict__ Cout, int M, int N, int K)
{
    __shared__ __bf16 As[128 * 64];   // [m][k]
    __shared__ __bf16 Bs[128 * 64];   // [n][k]

    const int tid  = threadIdx.x;
    const int wave = tid >> 5, lane = tid & 31;
    const int wm = wave >> 2, wn = wave & 3;          // 2 x 4 wave grid
    const int rowBase = blockIdx.y * 128;
    const int colBase = blockIdx.x * 128;

    const v8f vz = {0.f, 0.f, 0.f, 0.f, 0.f, 0.f, 0.f, 0.f};
    v8f acc[4][2];
#pragma unroll
    for (int i = 0; i < 4; ++i)
#pragma unroll
        for (int j = 0; j < 2; ++j) acc[i][j] = vz;

    for (int k0 = 0; k0 < K; k0 += 64) {
        for (int c = tid; c < 1024; c += 256) {        // A tile: 128x64
            int r = c >> 3, kg = c & 7;
            stage_b128(&A[(size_t)(rowBase + r) * K + k0 + kg * 8], &As[r * 64 + kg * 8]);
        }
        for (int c = tid; c < 1024; c += 256) {        // B tile: 128(n) x 64(k)
            int n = c >> 3, kg = c & 7;
            stage_b128(&Bt[(size_t)(colBase + n) * K + k0 + kg * 8], &Bs[n * 64 + kg * 8]);
        }
        async_join();
        __syncthreads();

#pragma unroll
        for (int kk = 0; kk < 64; kk += 32) {
            v16bf af[4], bf[2];
#pragma unroll
            for (int ms = 0; ms < 4; ++ms)
                af[ms] = ld_frag(&As[(size_t)(wm * 64 + ms * 16) * 64 + kk], 64, lane);
#pragma unroll
            for (int ns = 0; ns < 2; ++ns)
                bf[ns] = ld_frag(&Bs[(size_t)(wn * 32 + ns * 16) * 64 + kk], 64, lane);
#pragma unroll
            for (int ms = 0; ms < 4; ++ms)
#pragma unroll
                for (int ns = 0; ns < 2; ++ns)
                    acc[ms][ns] = WMMA_BF16(af[ms], bf[ns], acc[ms][ns]);
        }
        __syncthreads();
    }

    // Epilogue. C/D layout: element e of v8f -> row = e + 8*(lane/16), col = lane%16.
    const int hi = lane >> 4;
#pragma unroll
    for (int ms = 0; ms < 4; ++ms) {
#pragma unroll
        for (int ns = 0; ns < 2; ++ns) {
            const int row0 = rowBase + wm * 64 + ms * 16 + (hi << 3);
            const int col  = colBase + wn * 32 + ns * 16 + (lane & 15);
            const float bb = bias[col];
#pragma unroll
            for (int e = 0; e < 8; ++e) {
                const int row = row0 + e;
                float val = acc[ms][ns][e] + bb;
                if (MODE == EP_QKV) {
                    // col in [0,2304): which=q/k/v, head, d. Write [3,B,H,S,64] bf16,
                    // scaling Q by hd^-0.5 = 0.125 so attention consumes it directly.
                    int which = col / 768;
                    int rem = col - which * 768;
                    int hh = rem >> 6, d = rem & 63;
                    if (which == 0) val *= 0.125f;
                    size_t idx = (((size_t)(which * 48 + (row >> 11) * 12 + hh) << 11)
                                  + (size_t)(row & 2047)) * 64 + d;
                    ((__bf16*)Cout)[idx] = f2bf(val);
                } else if (MODE == EP_ATTN_RES) {
                    ((float*)Cout)[(size_t)row * N + col] = val + res[(size_t)row * N + col];
                } else if (MODE == EP_GELU) {
                    float gl = 0.5f * val * (1.0f + erff(val * 0.70710678118654752f));
                    ((__bf16*)Cout)[(size_t)row * N + col] = f2bf(gl);
                } else { // EP_OUT_RES
                    ((float*)Cout)[(size_t)row * N + col] = val + res[(size_t)row * N + col];
                }
            }
        }
    }
}

// ---------------------------------------------------------------------------
// Flash attention (no mask): one block per (b*h, 128 query rows).
// 8 waves, wave handles 16 query rows. KV tiles of 64. Q pre-scaled by 0.125.
// LDS: Qs 16K + Ks 8K + Vts 8K + Ps 16K = 48 KB (of 320 KB/WGP).
// ---------------------------------------------------------------------------
__global__ __launch_bounds__(256) void attention_kernel(const __bf16* __restrict__ qkv,
                                                        __bf16* __restrict__ o_out) {
    const int S = 2048, HD = 64, H = 12;
    __shared__ __bf16 Qs[128 * 64];    // [q][d]
    __shared__ __bf16 Ks[64 * 64];     // [kv][d]  (row-major == B-frag layout for Q.K^T)
    __shared__ __bf16 Vts[64 * 64];    // [d][kv]  (transposed == B-frag layout for P.V)
    __shared__ __bf16 Ps[128 * 64];    // [q][kv]  per-wave P re-layout (D-frag -> A-frag)

    const int tid = threadIdx.x, wave = tid >> 5, lane = tid & 31;
    const int bh = blockIdx.y;
    const int h  = bh % H;
    const int q0 = blockIdx.x * 128;

    const __bf16* Qp = qkv + (size_t)(bh)      * S * HD;   // which=0 (pre-scaled)
    const __bf16* Kp = qkv + (size_t)(48 + bh) * S * HD;   // which=1
    const __bf16* Vp = qkv + (size_t)(96 + bh) * S * HD;   // which=2

    for (int c = tid; c < 1024; c += 256) {
        int r = c >> 3, kg = c & 7;
        stage_b128(&Qp[(size_t)(q0 + r) * 64 + kg * 8], &Qs[r * 64 + kg * 8]);
    }

    const v8f vz = {0.f, 0.f, 0.f, 0.f, 0.f, 0.f, 0.f, 0.f};
    v8f o[4];
#pragma unroll
    for (int i = 0; i < 4; ++i) o[i] = vz;
    float mrow[8], lrow[8];
#pragma unroll
    for (int e = 0; e < 8; ++e) { mrow[e] = -1e30f; lrow[e] = 0.f; }

    async_join();
    __syncthreads();

    const int hi = lane >> 4;
    for (int kv0 = 0; kv0 < S; kv0 += 64) {
        {   // stage K rows (async contiguous) and V transposed (scatter)
            int r = tid >> 3, kg = tid & 7;                 // r in [0,32)
            stage_b128(&Kp[(size_t)(kv0 + r) * 64 + kg * 8],      &Ks[r * 64 + kg * 8]);
            stage_b128(&Kp[(size_t)(kv0 + 32 + r) * 64 + kg * 8], &Ks[(32 + r) * 64 + kg * 8]);
            uint4 raw0 = *(const uint4*)&Vp[(size_t)(kv0 + r) * 64 + kg * 8];
            uint4 raw1 = *(const uint4*)&Vp[(size_t)(kv0 + 32 + r) * 64 + kg * 8];
            const __bf16* v0p = (const __bf16*)&raw0;
            const __bf16* v1p = (const __bf16*)&raw1;
#pragma unroll
            for (int j = 0; j < 8; ++j) {
                Vts[(kg * 8 + j) * 64 + r]      = v0p[j];
                Vts[(kg * 8 + j) * 64 + 32 + r] = v1p[j];
            }
        }
        async_join();
        __syncthreads();

        // scores S = (Q*scale) . K^T : 16 x 64 per wave
        v8f sc[4] = {vz, vz, vz, vz};
#pragma unroll
        for (int kk = 0; kk < 64; kk += 32) {
            v16bf aq = ld_frag(&Qs[(size_t)(wave * 16) * 64 + kk], 64, lane);
#pragma unroll
            for (int ns = 0; ns < 4; ++ns) {
                v16bf bk = ld_frag(&Ks[(size_t)(ns * 16) * 64 + kk], 64, lane);
                sc[ns] = WMMA_BF16(aq, bk, sc[ns]);
            }
        }

        // online softmax: row r = e + 8*hi lives across 16 lanes of one half-wave
#pragma unroll
        for (int e = 0; e < 8; ++e) {
            float v0 = sc[0][e], v1 = sc[1][e], v2 = sc[2][e], v3 = sc[3][e];
            float mx = fmaxf(fmaxf(v0, v1), fmaxf(v2, v3));
#pragma unroll
            for (int off = 1; off < 16; off <<= 1)
                mx = fmaxf(mx, __shfl_xor(mx, off, 16));
            float nm = fmaxf(mrow[e], mx);
            float alpha = __expf(mrow[e] - nm);
            float p0 = __expf(v0 - nm), p1 = __expf(v1 - nm);
            float p2 = __expf(v2 - nm), p3 = __expf(v3 - nm);
            float rs = (p0 + p1) + (p2 + p3);
#pragma unroll
            for (int off = 1; off < 16; off <<= 1)
                rs += __shfl_xor(rs, off, 16);
            lrow[e] = lrow[e] * alpha + rs;
            mrow[e] = nm;
#pragma unroll
            for (int d4 = 0; d4 < 4; ++d4) o[d4][e] *= alpha;
            int prow = wave * 16 + e + (hi << 3);
            Ps[prow * 64 + (lane & 15)]      = f2bf(p0);
            Ps[prow * 64 + 16 + (lane & 15)] = f2bf(p1);
            Ps[prow * 64 + 32 + (lane & 15)] = f2bf(p2);
            Ps[prow * 64 + 48 + (lane & 15)] = f2bf(p3);
        }

        // O += P . V   (K=64 -> 2 WMMA per 16-wide d subtile)
#pragma unroll
        for (int kk = 0; kk < 64; kk += 32) {
            v16bf ap = ld_frag(&Ps[(size_t)(wave * 16) * 64 + kk], 64, lane);
#pragma unroll
            for (int d4 = 0; d4 < 4; ++d4) {
                v16bf bv = ld_frag(&Vts[(size_t)(d4 * 16) * 64 + kk], 64, lane);
                o[d4] = WMMA_BF16(ap, bv, o[d4]);
            }
        }
        __syncthreads();
    }

    // write O / l  -> [B*S, 768] bf16 at column h*64
#pragma unroll
    for (int d4 = 0; d4 < 4; ++d4)
#pragma unroll
        for (int e = 0; e < 8; ++e) {
            int row = q0 + wave * 16 + e + (hi << 3);
            int col = h * 64 + d4 * 16 + (lane & 15);
            o_out[(size_t)row * 768 + col] = f2bf(o[d4][e] * (1.0f / lrow[e]));
        }
}

// ---------------------------------------------------------------------------
// Launch
// ---------------------------------------------------------------------------
extern "C" void kernel_launch(void* const* d_in, const int* in_sizes, int n_in,
                              void* d_out, int out_size, void* d_ws, size_t ws_size,
                              hipStream_t stream) {
    (void)in_sizes; (void)n_in; (void)out_size; (void)ws_size;
    const int B = 4, S = 2048, D = 768, DFF = 3072;
    const int M = B * S;                       // 8192

    const float* hidden = (const float*)d_in[0];
    const float* ln1_g  = (const float*)d_in[1];
    const float* ln1_b  = (const float*)d_in[2];
    const float* w_qkv  = (const float*)d_in[3];
    const float* b_qkv  = (const float*)d_in[4];
    const float* w_attn = (const float*)d_in[5];
    const float* b_attn = (const float*)d_in[6];
    const float* ln2_g  = (const float*)d_in[7];
    const float* ln2_b  = (const float*)d_in[8];
    const float* w_in   = (const float*)d_in[9];
    const float* b_in   = (const float*)d_in[10];
    const float* w_out  = (const float*)d_in[11];
    const float* b_out  = (const float*)d_in[12];

    // workspace carve-up (bytes, all 256-aligned by construction)
    char* ws = (char*)d_ws;
    __bf16* wqkv_t  = (__bf16*)(ws);                                   // [2304][768]
    __bf16* wattn_t = (__bf16*)(ws + 3538944);                         // [768][768]
    __bf16* win_t   = (__bf16*)(ws + 4718592);                         // [3072][768]
    __bf16* wout_t  = (__bf16*)(ws + 9437184);                         // [768][3072]
    __bf16* ln1_bf  = (__bf16*)(ws + 14155776);                        // 8192*768
    __bf16* qkv_bf  = (__bf16*)(ws + 26738688);                        // 3*4*12*2048*64
    __bf16* attno_bf= (__bf16*)(ws + 64487424);                        // 8192*768
    float*  h_f32   = (float*) (ws + 77070336);                        // 8192*768 f32
    __bf16* ln2_bf  = (__bf16*)(ws + 102236160);                       // 8192*768
    __bf16* ff_bf   = (__bf16*)(ws + 114819072);                       // 8192*3072

    // 1) weights -> bf16, transposed to [N][K]
    transpose_bf16_kernel<<<dim3(3 * D / 32, D / 32), 256, 0, stream>>>(w_qkv, wqkv_t, D, 3 * D);
    transpose_bf16_kernel<<<dim3(D / 32, D / 32), 256, 0, stream>>>(w_attn, wattn_t, D, D);
    transpose_bf16_kernel<<<dim3(DFF / 32, D / 32), 256, 0, stream>>>(w_in, win_t, D, DFF);
    transpose_bf16_kernel<<<dim3(D / 32, DFF / 32), 256, 0, stream>>>(w_out, wout_t, DFF, D);

    // 2) LN1
    layernorm_kernel<<<M, 256, 0, stream>>>(hidden, ln1_g, ln1_b, ln1_bf, D);

    // 3) QKV GEMM (8192 x 2304 x 768), epilogue scatters into [3,B,H,S,64] + scales Q
    gemm_bf16_kernel<EP_QKV><<<dim3((3 * D) / 128, M / 128), 256, 0, stream>>>(
        ln1_bf, wqkv_t, b_qkv, nullptr, (void*)qkv_bf, M, 3 * D, D);

    // 4) flash attention -> [8192, 768] bf16
    attention_kernel<<<dim3(S / 128, B * 12), 256, 0, stream>>>(qkv_bf, attno_bf);

    // 5) attention projection + residual -> h (f32)
    gemm_bf16_kernel<EP_ATTN_RES><<<dim3(D / 128, M / 128), 256, 0, stream>>>(
        attno_bf, wattn_t, b_attn, hidden, (void*)h_f32, M, D, D);

    // 6) LN2
    layernorm_kernel<<<M, 256, 0, stream>>>(h_f32, ln2_g, ln2_b, ln2_bf, D);

    // 7) FF-in GEMM + exact GELU -> bf16
    gemm_bf16_kernel<EP_GELU><<<dim3(DFF / 128, M / 128), 256, 0, stream>>>(
        ln2_bf, win_t, b_in, nullptr, (void*)ff_bf, M, DFF, D);

    // 8) FF-out GEMM + bias + residual2 -> f32 output
    gemm_bf16_kernel<EP_OUT_RES><<<dim3(D / 128, M / 128), 256, 0, stream>>>(
        ff_bf, wout_t, b_out, h_f32, d_out, M, D, DFF);
}
